// MambaBlock_70995809402931
// MI455X (gfx1250) — matrix operand order repaired and checked by reading, not compiled, not run on hardware
//
#include <hip/hip_runtime.h>

// ============================================================================
// Mamba block forward for MI455X (gfx1250, wave32, WMMA + TDM).
//
// Pipeline (all on `stream`, scratch carved from d_ws):
//   1) f32->bf16 weight conversion (in_proj, out_proj)
//   2) RMSNorm -> xn (bf16)
//   3) WMMA GEMM: xz = xn @ in_proj_w^T          [16384 x 4096] bf16 out
//   4) depthwise causal conv(K=4) + SiLU -> u    [16384 x 2048] f32
//   5) x_proj:  x_dbl = u @ x_proj_w^T           [16384 x 96]   f32
//   6) dt_proj: delta = softplus(dt @ W^T + b)   [16384 x 2048] f32
//   7) selective scan (1 lane per (b,d) channel) -> y*silu(z)  bf16
//   8) WMMA GEMM: out = y @ out_proj_w^T + residual(x)         f32
//
// GEMM: v_wmma_f32_16x16x32_bf16, block tile 128x128, 8 waves of 2x4.
// Tile staging uses the Tensor Data Mover (tensor_load_to_lds +
// s_wait_tensorcnt) when the builtin is available; the D# pad fields
// (4 DWORDs pad per 16 DWORDs) implement the 40-element LDS row stride.
// ============================================================================

typedef __bf16 bf16;
typedef bf16  v16bf __attribute__((ext_vector_type(16)));
typedef bf16  v8bf  __attribute__((ext_vector_type(8)));
typedef float v8f   __attribute__((ext_vector_type(8)));
typedef unsigned int u32x4 __attribute__((ext_vector_type(4)));
typedef int i32x4 __attribute__((ext_vector_type(4)));
typedef int i32x8 __attribute__((ext_vector_type(8)));

#define BATCH   8
#define SEQ     2048
#define DMODEL  1024
#define DINNER  2048
#define DTRANK  64
#define DSTATE  16
#define MTOT    (BATCH * SEQ)        // 16384 rows
#define LDS_STRIDE 40                // 32 data + 8 pad bf16 per LDS tile row
#define TILE_BYTES (128 * LDS_STRIDE * 2)   // 10240 B per tile buffer

#if __has_builtin(__builtin_amdgcn_tensor_load_to_lds) && \
    __has_builtin(__builtin_amdgcn_s_wait_tensorcnt)
#define MAMBA_USE_TDM 1
#else
#define MAMBA_USE_TDM 0
#endif

// ---------------------------------------------------------------------------
// f32 -> bf16 elementwise (weight conversion)
// ---------------------------------------------------------------------------
__global__ __launch_bounds__(256)
void mamba_cvt_bf16(const float* __restrict__ in, bf16* __restrict__ out, long n) {
    long i = (long)blockIdx.x * 256 + threadIdx.x;
    if (i < n) out[i] = (bf16)in[i];
}

// ---------------------------------------------------------------------------
// RMSNorm: one block per row of 1024, writes bf16
// ---------------------------------------------------------------------------
__global__ __launch_bounds__(256)
void mamba_rmsnorm(const float* __restrict__ x, const float* __restrict__ w,
                   bf16* __restrict__ xn) {
    __shared__ float red[256];
    const long row = blockIdx.x;
    const float* xr = x + row * DMODEL;
    float s = 0.f;
    for (int i = threadIdx.x; i < DMODEL; i += 256) { float v = xr[i]; s += v * v; }
    red[threadIdx.x] = s;
    __syncthreads();
    for (int off = 128; off > 0; off >>= 1) {
        if (threadIdx.x < off) red[threadIdx.x] += red[threadIdx.x + off];
        __syncthreads();
    }
    const float scale = rsqrtf(red[0] * (1.0f / DMODEL) + 1e-5f);
    bf16* o = xn + row * DMODEL;
    for (int i = threadIdx.x; i < DMODEL; i += 256)
        o[i] = (bf16)(xr[i] * scale * w[i]);
}

#if MAMBA_USE_TDM
// ---------------------------------------------------------------------------
// Issue a TDM 2D tile load: 128 rows x 32 bf16 from row-major global (leading
// dim = strideElems) into LDS at byte offset ldsOff, with 4-DWORD padding
// every 16 DWORDs (-> LDS_STRIDE=40 bf16 row pitch). Wave-level instruction.
// D# layout per CDNA5 ISA ch. 8 (group0 128b, group1 256b; groups 2/3 unused
// for 2D tensors).
// ---------------------------------------------------------------------------
__device__ __forceinline__ void mamba_tdm_load_tile(unsigned ldsOff,
                                                    const bf16* gptr,
                                                    unsigned strideElems) {
    const unsigned long long ga = (unsigned long long)(uintptr_t)gptr;
    // group 0: count=1 | lds_addr | global_addr[56:0] | type=2
    unsigned g0w0 = 1u;
    unsigned g0w1 = ldsOff;
    unsigned g0w2 = (unsigned)(ga & 0xFFFFFFFFu);
    unsigned g0w3 = (unsigned)((ga >> 32) & 0x01FFFFFFu) | (2u << 30);
    // group 1: data_size=2B(1), pad_enable, pad_interval=16dw(3), pad_amount=4dw(3)
    unsigned g1w0 = (1u << 16) | (1u << 20) | (3u << 22) | (3u << 25);
    unsigned g1w1 = (32u << 16);            // tensor_dim0[15:0]  = 32
    unsigned g1w2 = (128u << 16);           // tensor_dim1[15:0]  = 128
    unsigned g1w3 = (32u << 16);            // tile_dim0          = 32
    unsigned g1w4 = 128u;                   // tile_dim1 = 128, tile_dim2 = 0
    unsigned g1w5 = strideElems;            // tensor_dim0_stride[31:0]
    unsigned g1w6 = 0u;
    unsigned g1w7 = 0u;
    // Force descriptor dwords into SGPRs.
    u32x4 g0;
    g0.x = (unsigned)__builtin_amdgcn_readfirstlane((int)g0w0);
    g0.y = (unsigned)__builtin_amdgcn_readfirstlane((int)g0w1);
    g0.z = (unsigned)__builtin_amdgcn_readfirstlane((int)g0w2);
    g0.w = (unsigned)__builtin_amdgcn_readfirstlane((int)g0w3);
    i32x8 g1;
    g1[0] = __builtin_amdgcn_readfirstlane((int)g1w0);
    g1[1] = __builtin_amdgcn_readfirstlane((int)g1w1);
    g1[2] = __builtin_amdgcn_readfirstlane((int)g1w2);
    g1[3] = __builtin_amdgcn_readfirstlane((int)g1w3);
    g1[4] = __builtin_amdgcn_readfirstlane((int)g1w4);
    g1[5] = __builtin_amdgcn_readfirstlane((int)g1w5);
    g1[6] = __builtin_amdgcn_readfirstlane((int)g1w6);
    g1[7] = __builtin_amdgcn_readfirstlane((int)g1w7);
#if defined(__clang_major__) && (__clang_major__ >= 23)
    __builtin_amdgcn_tensor_load_to_lds(g0, g1, (i32x4)0, (i32x4)0, (i32x8)0, 0);
#else
    __builtin_amdgcn_tensor_load_to_lds(g0, g1, (i32x4)0, (i32x4)0, 0);
#endif
}
#endif // MAMBA_USE_TDM

// ---------------------------------------------------------------------------
// WMMA GEMM: C[M,N] = A[M,K] @ W[N,K]^T   (A, W bf16 row-major)
// MODE 0: store bf16 into Cbf.  MODE 1: store f32 (acc + Res) into Cf.
// Grid: (N/128, M/128), block 256 (8 waves), dynamic LDS = 2*TILE_BYTES.
// A/B fragment gathers follow CDNA5 wave32 16-bit WMMA VGPR layouts:
//   A 16x32: lane<16 -> m=lane, elems = K{0..7},{16..23};
//            lane>=16 -> m=lane-16, elems = K{8..15},{24..31}.
//   B 32x16 staged in W layout [n][k]: lane<16 -> n=lane, elems=K{0..7},{16..23};
//            lane>=16 -> n=lane-16, elems=K{8..15},{24..31}.
// ---------------------------------------------------------------------------
template <int MODE>
__global__ __launch_bounds__(256)
void mamba_wmma_gemm(const bf16* __restrict__ A, int lda,
                     const bf16* __restrict__ W, int ldb,
                     bf16* __restrict__ Cbf, float* __restrict__ Cf,
                     const float* __restrict__ Res,
                     int K, int ldc) {
    extern __shared__ __align__(16) bf16 smem[];
    bf16* sA = smem;                          // LDS byte offset 0
    bf16* sB = smem + 128 * LDS_STRIDE;       // LDS byte offset TILE_BYTES

    const int t     = threadIdx.x;
    const int lane  = t & 31;
    const int wid   = t >> 5;        // 0..7
    const int waveM = wid >> 2;      // 0..1  (rows: 64 each)
    const int waveN = wid & 3;       // 0..3  (cols: 32 each)

    const long bm = (long)blockIdx.y * 128;
    const long bn = (long)blockIdx.x * 128;

    // fallback staging coords: each thread copies 16 bf16 (two 16B chunks)
    const int srow  = t >> 1;        // 0..127
    const int shalf = (t & 1) * 16;  // 0 or 16
    const long aBase = (bm + srow) * (long)lda + shalf;
    const long bBase = (bn + srow) * (long)ldb + shalf;

    const int half = lane >> 4;      // 0/1
    const int lr   = lane & 15;

    v8f acc[4][2];
#pragma unroll
    for (int i = 0; i < 4; ++i)
#pragma unroll
        for (int j = 0; j < 2; ++j) acc[i][j] = (v8f)0.f;

    const int nkt = K >> 5;
    for (int kt = 0; kt < nkt; ++kt) {
        const int kOff = kt << 5;
        if (kt + 1 < nkt) {          // prefetch next K tile -> global_prefetch_b8
            __builtin_prefetch(A + aBase + kOff + 32, 0, 1);
            __builtin_prefetch(W + bBase + kOff + 32, 0, 1);
        }
#if MAMBA_USE_TDM
        __syncthreads();             // previous K-tile fully consumed
        if (wid == 0) {              // one wave drives the TDM
            mamba_tdm_load_tile(0u,          A + bm * (long)lda + kOff, (unsigned)lda);
            mamba_tdm_load_tile(TILE_BYTES,  W + bn * (long)ldb + kOff, (unsigned)ldb);
            __builtin_amdgcn_s_wait_tensorcnt(0);
        }
        __syncthreads();             // tiles visible to all waves
#else
        // ---- synchronous staging fallback ----
        const uint4* gA = (const uint4*)(A + aBase + kOff);
        const uint4* gB = (const uint4*)(W + bBase + kOff);
        uint4 ra0 = gA[0], ra1 = gA[1];
        uint4 rb0 = gB[0], rb1 = gB[1];
        uint4* lA = (uint4*)(sA + srow * LDS_STRIDE + shalf);
        uint4* lB = (uint4*)(sB + srow * LDS_STRIDE + shalf);
        lA[0] = ra0; lA[1] = ra1;
        lB[0] = rb0; lB[1] = rb1;
        __syncthreads();
#endif

        // ---- gather fragments per CDNA5 layouts ----
        v16bf aFrag[4];
#pragma unroll
        for (int i = 0; i < 4; ++i) {
            const bf16* p = sA + (waveM * 64 + i * 16 + lr) * LDS_STRIDE;
            v8bf lo = *(const v8bf*)(p + half * 8);
            v8bf hi = *(const v8bf*)(p + 16 + half * 8);
            aFrag[i] = __builtin_shufflevector(lo, hi,
                0, 1, 2, 3, 4, 5, 6, 7, 8, 9, 10, 11, 12, 13, 14, 15);
        }
        v16bf bFrag[2];
#pragma unroll
        for (int j = 0; j < 2; ++j) {
            const bf16* p = sB + (waveN * 32 + j * 16 + lr) * LDS_STRIDE;
            v8bf lo = *(const v8bf*)(p + half * 8);
            v8bf hi = *(const v8bf*)(p + 16 + half * 8);
            bFrag[j] = __builtin_shufflevector(lo, hi,
                0, 1, 2, 3, 4, 5, 6, 7, 8, 9, 10, 11, 12, 13, 14, 15);
        }

        // ---- 8 WMMAs per wave per K-tile ----
#pragma unroll
        for (int i = 0; i < 4; ++i)
#pragma unroll
            for (int j = 0; j < 2; ++j)
                acc[i][j] = __builtin_amdgcn_wmma_f32_16x16x32_bf16(
                    false, aFrag[i], false, bFrag[j],
                    (short)0, acc[i][j], false, false);
#if !MAMBA_USE_TDM
        __syncthreads();
#endif
    }

    // ---- epilogue: D tile layout: VGPR r -> m = r + 8*half, n = lane&15 ----
#pragma unroll
    for (int i = 0; i < 4; ++i) {
#pragma unroll
        for (int j = 0; j < 2; ++j) {
            const long row0 = bm + waveM * 64 + i * 16 + half * 8;
            const long col  = bn + waveN * 32 + j * 16 + lr;
#pragma unroll
            for (int r = 0; r < 8; ++r) {
                const long idx = (row0 + r) * (long)ldc + col;
                if (MODE == 0) {
                    Cbf[idx] = (bf16)acc[i][j][r];
                } else {
                    Cf[idx] = acc[i][j][r] + Res[idx];
                }
            }
        }
    }
}

// ---------------------------------------------------------------------------
// Depthwise causal conv (K=4) + SiLU.  xz bf16 [MTOT x 4096], x_part = cols<2048.
// ---------------------------------------------------------------------------
__global__ __launch_bounds__(256)
void mamba_conv_silu(const bf16* __restrict__ xz, const float* __restrict__ cw,
                     const float* __restrict__ cb, float* __restrict__ u) {
    const long idx = (long)blockIdx.x * 256 + threadIdx.x;   // over MTOT*DINNER
    const int  d = (int)(idx & (DINNER - 1));
    const long m = idx >> 11;
    const int  l = (int)(m & (SEQ - 1));
    float acc = cb[d];
#pragma unroll
    for (int j = 0; j < 4; ++j) {
        const int ls = l + j - 3;
        if (ls >= 0)
            acc += cw[d * 4 + j] * (float)xz[(m - 3 + j) * (2 * DINNER) + d];
    }
    u[idx] = acc / (1.f + __expf(-acc));   // SiLU
}

// ---------------------------------------------------------------------------
// x_proj: x_dbl[m, c] = sum_d u[m,d] * Wx[c,d],  c < 96. One block per row m.
// ---------------------------------------------------------------------------
__global__ __launch_bounds__(128)
void mamba_xproj(const float* __restrict__ u, const float* __restrict__ wx,
                 float* __restrict__ xdbl) {
    __shared__ float sU[DINNER];
    const long m = blockIdx.x;
    for (int i = threadIdx.x; i < DINNER; i += 128) sU[i] = u[m * DINNER + i];
    __syncthreads();
    const int c = threadIdx.x;
    if (c < DTRANK + 2 * DSTATE) {
        const float* wr = wx + (long)c * DINNER;
        float acc = 0.f;
        for (int d = 0; d < DINNER; ++d) acc += wr[d] * sU[d];
        xdbl[m * 96 + c] = acc;
    }
}

// ---------------------------------------------------------------------------
// dt_proj + softplus: delta[m,d] = softplus(dot64(xdbl[m,:64], Wdt[d,:]) + b[d])
// ---------------------------------------------------------------------------
__global__ __launch_bounds__(256)
void mamba_dtproj(const float* __restrict__ xdbl, const float* __restrict__ wdt,
                  const float* __restrict__ bdt, float* __restrict__ delta) {
    const long idx = (long)blockIdx.x * 256 + threadIdx.x;
    const int  d = (int)(idx & (DINNER - 1));
    const long m = idx >> 11;
    const float* xr = xdbl + m * 96;
    const float* wr = wdt + (long)d * DTRANK;
    float acc = bdt[d];
#pragma unroll 8
    for (int r = 0; r < DTRANK; ++r) acc += xr[r] * wr[r];
    delta[idx] = (acc > 20.f) ? acc : __logf(1.f + __expf(acc));
}

// ---------------------------------------------------------------------------
// Selective scan: one lane per (b, d) channel; 16-state recurrence over L.
// Fuses output gating y *= silu(z) and bf16 store for the out_proj GEMM.
// ---------------------------------------------------------------------------
__global__ __launch_bounds__(256)
void mamba_scan(const float* __restrict__ u, const float* __restrict__ delta,
                const float* __restrict__ xdbl, const bf16* __restrict__ xz,
                const float* __restrict__ A_log, const float* __restrict__ Dv,
                bf16* __restrict__ y) {
    const int idx = blockIdx.x * 256 + threadIdx.x;  // 0 .. 16383
    const int d = idx & (DINNER - 1);
    const int b = idx >> 11;
    float a[DSTATE], h[DSTATE];
#pragma unroll
    for (int n = 0; n < DSTATE; ++n) {
        a[n] = -__expf(A_log[d * DSTATE + n]);
        h[n] = 0.f;
    }
    const float Dd = Dv[d];
    const long mBase = (long)b * SEQ;
    for (int t = 0; t < SEQ; ++t) {
        const long m = mBase + t;
        const float dt = delta[m * DINNER + d];
        const float ut = u[m * DINNER + d];
        const float dtu = dt * ut;
        const float* bc = xdbl + m * 96 + DTRANK;    // B then C
        float yt = 0.f;
#pragma unroll
        for (int n = 0; n < DSTATE; ++n) {
            const float dA = __expf(dt * a[n]);
            h[n] = dA * h[n] + dtu * bc[n];
            yt += h[n] * bc[DSTATE + n];
        }
        yt += ut * Dd;
        const float z = (float)xz[m * (2 * DINNER) + DINNER + d];
        yt *= z / (1.f + __expf(-z));                // y * silu(z)
        y[m * DINNER + d] = (bf16)yt;
    }
}

// ---------------------------------------------------------------------------
// Host-side launcher
// ---------------------------------------------------------------------------
extern "C" void kernel_launch(void* const* d_in, const int* in_sizes, int n_in,
                              void* d_out, int out_size, void* d_ws, size_t ws_size,
                              hipStream_t stream) {
    const float* x        = (const float*)d_in[0];
    const float* norm_w   = (const float*)d_in[1];
    const float* in_projw = (const float*)d_in[2];
    const float* conv_w   = (const float*)d_in[3];
    const float* conv_b   = (const float*)d_in[4];
    const float* x_projw  = (const float*)d_in[5];
    const float* dt_projw = (const float*)d_in[6];
    const float* dt_projb = (const float*)d_in[7];
    const float* A_log    = (const float*)d_in[8];
    const float* Dvec     = (const float*)d_in[9];
    const float* out_projw= (const float*)d_in[10];
    float* out = (float*)d_out;

    // ---- workspace layout (all sizes multiples of 256 B) ----
    char* ws = (char*)d_ws;
    size_t off = 0;
    bf16* wIn  = (bf16*)(ws + off); off += (size_t)2 * DINNER * DMODEL * sizeof(bf16); // 8 MB
    bf16* wOut = (bf16*)(ws + off); off += (size_t)DMODEL * DINNER * sizeof(bf16);     // 4 MB
    bf16* xn   = (bf16*)(ws + off); off += (size_t)MTOT * DMODEL * sizeof(bf16);       // 32 MB
    bf16* xz   = (bf16*)(ws + off); off += (size_t)MTOT * 2 * DINNER * sizeof(bf16);   // 128 MB
    float* u   = (float*)(ws + off); off += (size_t)MTOT * DINNER * sizeof(float);     // 128 MB
    float* xdbl= (float*)(ws + off); off += (size_t)MTOT * 96 * sizeof(float);         // 6 MB
    float* dlt = (float*)(ws + off); off += (size_t)MTOT * DINNER * sizeof(float);     // 128 MB
    bf16* ybuf = (bf16*)(ws + off); off += (size_t)MTOT * DINNER * sizeof(bf16);       // 64 MB

    // 1) weight conversion
    {
        long n1 = (long)2 * DINNER * DMODEL;   // 4,194,304
        long n2 = (long)DMODEL * DINNER;       // 2,097,152
        mamba_cvt_bf16<<<(unsigned)((n1 + 255) / 256), 256, 0, stream>>>(in_projw, wIn, n1);
        mamba_cvt_bf16<<<(unsigned)((n2 + 255) / 256), 256, 0, stream>>>(out_projw, wOut, n2);
    }

    // 2) RMSNorm
    mamba_rmsnorm<<<MTOT, 256, 0, stream>>>(x, norm_w, xn);

    // 3) in_proj GEMM: [16384 x 1024] x [4096 x 1024]^T -> xz bf16 [16384 x 4096]
    {
        dim3 grid((2 * DINNER) / 128, MTOT / 128);   // (32, 128)
        mamba_wmma_gemm<0><<<grid, 256, 2 * TILE_BYTES, stream>>>(
            xn, DMODEL, wIn, DMODEL, xz, (float*)nullptr, (const float*)nullptr,
            DMODEL, 2 * DINNER);
    }

    // 4) conv + SiLU
    mamba_conv_silu<<<(MTOT * DINNER) / 256, 256, 0, stream>>>(xz, conv_w, conv_b, u);

    // 5) x_proj
    mamba_xproj<<<MTOT, 128, 0, stream>>>(u, x_projw, xdbl);

    // 6) dt_proj + softplus
    mamba_dtproj<<<(MTOT * DINNER) / 256, 256, 0, stream>>>(xdbl, dt_projw, dt_projb, dlt);

    // 7) selective scan + gating
    mamba_scan<<<(BATCH * DINNER) / 256, 256, 0, stream>>>(u, dlt, xdbl, xz, A_log, Dvec, ybuf);

    // 8) out_proj GEMM + residual: [16384 x 2048] x [1024 x 2048]^T + x -> out f32
    {
        dim3 grid(DMODEL / 128, MTOT / 128);         // (8, 128)
        mamba_wmma_gemm<1><<<grid, 256, 2 * TILE_BYTES, stream>>>(
            ybuf, DINNER, wOut, DINNER, (bf16*)nullptr, out, x,
            DINNER, DMODEL);
    }
    (void)in_sizes; (void)n_in; (void)out_size; (void)ws_size;
}
